// MultiHeadAttention_78211354460614
// MI455X (gfx1250) — matrix operand order repaired and checked by reading, not compiled, not run on hardware
//
#include <hip/hip_runtime.h>
#include <hip/hip_bf16.h>

// ---------------------------------------------------------------------------
// MHA forward for MI455X (gfx1250), wave32 + v_wmma_f32_16x16x32_bf16.
// Round 3: K tiles in the attention kernel are now streamed by the Tensor
// Data Mover (tensor_load_to_lds, TENSORcnt double-buffering) instead of
// per-lane VMEM gathers. Everything else: bf16 pre-staging, b128 fragments,
// Vt per-head transpose, P re-layout via LDS.
// ---------------------------------------------------------------------------

#define BATCH 4
#define NHEAD 16
#define S_LEN 2048
#define DM    1024
#define HD    64
#define LDS_P 40   // padded LDS row stride (ushorts): 80B rows, 16B-aligned frags

// LDS map (ushort indices): [0,2560) P tiles; [2560,4608) K buf0; [4608,6656) K buf1
#define LDSU_P    0
#define LDSU_K0   2560
#define LDSU_K1   4608
#define LDSB_K0   (LDSU_K0 * 2)   // byte offsets for TDM descriptors
#define LDSB_K1   (LDSU_K1 * 2)

typedef __attribute__((ext_vector_type(16))) __bf16 v16bf;
typedef __attribute__((ext_vector_type(8)))  float  v8f;
typedef __attribute__((ext_vector_type(4)))  unsigned su4;
typedef __attribute__((ext_vector_type(8)))  unsigned su8;

struct alignas(16) u32x4 { unsigned x, y, z, w; };

union BF16x16 {
    unsigned short u[16];
    u32x4          q[2];
    v16bf          v;
};

__device__ __forceinline__ unsigned short f2bfu(float f) {
    unsigned u = __float_as_uint(f);
    u += 0x7fffu + ((u >> 16) & 1u);   // round-to-nearest-even
    return (unsigned short)(u >> 16);
}

// ---------------------------------------------------------------------------
// TDM: DMA a 32(keys) x 64(dims) bf16 tile from the K staging tensor into LDS.
// D# group0: count=1, lds_addr, global_addr(tile start), type=2.
// D# group1: data_size=2B, tensor_dim0=1024, tensor_dim1=8192,
//            tile_dim0=64, tile_dim1=32, dim0_stride=1024.
// ---------------------------------------------------------------------------
__device__ __forceinline__ void tdm_load_k_tile(const unsigned short* gptr,
                                                unsigned lds_byte_off) {
    const unsigned long long ga = (unsigned long long)(uintptr_t)gptr;
    su4 g0;
    g0[0] = 1u;                                            // count=1, user mode
    g0[1] = lds_byte_off;                                  // lds_addr (bytes)
    g0[2] = (unsigned)ga;                                  // global_addr[31:0]
    g0[3] = (unsigned)((ga >> 32) & 0x01FFFFFFu)           // global_addr[56:32]
          | (2u << 30);                                    // type=2 ("image")
    su8 g1;
    g1[0] = 0x00010000u;                                   // data_size=2B
    g1[1] = ((unsigned)DM & 0xFFFFu) << 16;                // tensor_dim0 lo16
    g1[2] = ((unsigned)DM >> 16)
          | (((unsigned)(BATCH * S_LEN) & 0xFFFFu) << 16); // dim0 hi / dim1 lo
    g1[3] = ((unsigned)(BATCH * S_LEN) >> 16)
          | (64u << 16);                                   // dim1 hi / tile_dim0
    g1[4] = 32u;                                           // tile_dim1 (tile_dim2=0)
    g1[5] = (unsigned)DM;                                  // dim0_stride lo32
    g1[6] = 0u;                                            // stride hi / dim1_stride
    g1[7] = 0u;
    asm volatile("tensor_load_to_lds %0, %1" :: "s"(g0), "s"(g1) : "memory");
}

// ---------------------------------------------------------------------------
// Elementwise f32 -> bf16 (packed, 8 elems/thread, b128 in / b128 out)
// ---------------------------------------------------------------------------
__global__ __launch_bounds__(256)
void k_cvt_bf16(const float* __restrict__ X, unsigned short* __restrict__ Y, int n) {
    const int i = (blockIdx.x * 256 + threadIdx.x) * 8;
    if (i >= n) return;
    float4 a = *(const float4*)(X + i);
    float4 b = *(const float4*)(X + i + 4);
    u32x4 o;
    o.x = (unsigned)f2bfu(a.x) | ((unsigned)f2bfu(a.y) << 16);
    o.y = (unsigned)f2bfu(a.z) | ((unsigned)f2bfu(a.w) << 16);
    o.z = (unsigned)f2bfu(b.x) | ((unsigned)f2bfu(b.y) << 16);
    o.w = (unsigned)f2bfu(b.z) | ((unsigned)f2bfu(b.w) << 16);
    *(u32x4*)(Y + i) = o;
}

// ---------------------------------------------------------------------------
// GEMM: Y = Xb[M,K](bf16) @ Wb[N,K]^T (bf16) + bias
// One 64x16 C tile per wave (4 WMMAs per B-fragment load).
// mode 0: bf16 row-major; mode 1: bf16 transposed per head Vt[b,h,dh,S];
// mode 2: f32 row-major.
// ---------------------------------------------------------------------------
__global__ __launch_bounds__(256)
void k_gemm_bf16(const unsigned short* __restrict__ Xb,
                 const unsigned short* __restrict__ Wb,
                 const float* __restrict__ bias,
                 void* __restrict__ Yraw,
                 int M, int N, int K, int mode) {
    const int lane = threadIdx.x & 31;
    const int wid  = threadIdx.x >> 5;
    const int g    = lane >> 4;
    const int l15  = lane & 15;
    const int ntiles = N >> 4;
    const int mtiles = M >> 6;
    const int tile = blockIdx.x * 8 + wid;
    if (tile >= mtiles * ntiles) return;          // wave-uniform
    const int m0 = (tile / ntiles) << 6;
    const int n0 = (tile % ntiles) << 4;

    const unsigned short* x0   = Xb + (size_t)(m0 + l15) * K;
    const unsigned short* wrow = Wb + (size_t)(n0 + l15) * K;

    v8f acc[4] = {{}, {}, {}, {}};
    for (int k0 = 0; k0 < K; k0 += 32) {
        __builtin_prefetch(wrow + k0 + 512, 0, 1);     // global_prefetch_b8
        BF16x16 bm;                                    // B: interleaved halves, as A
        bm.q[0] = *(const u32x4*)(wrow + k0 + (g << 3));
        bm.q[1] = *(const u32x4*)(wrow + k0 + (g << 3) + 16);
        const int ha = k0 + (g << 3);                  // A: interleaved halves
        BF16x16 a0, a1, a2, a3;
        a0.q[0] = *(const u32x4*)(x0 + ha);
        a0.q[1] = *(const u32x4*)(x0 + ha + 16);
        a1.q[0] = *(const u32x4*)(x0 + 16 * K + ha);
        a1.q[1] = *(const u32x4*)(x0 + 16 * K + ha + 16);
        a2.q[0] = *(const u32x4*)(x0 + 32 * K + ha);
        a2.q[1] = *(const u32x4*)(x0 + 32 * K + ha + 16);
        a3.q[0] = *(const u32x4*)(x0 + 48 * K + ha);
        a3.q[1] = *(const u32x4*)(x0 + 48 * K + ha + 16);
        acc[0] = __builtin_amdgcn_wmma_f32_16x16x32_bf16(false, a0.v, false, bm.v,
                                                         (short)0, acc[0], false, false);
        acc[1] = __builtin_amdgcn_wmma_f32_16x16x32_bf16(false, a1.v, false, bm.v,
                                                         (short)0, acc[1], false, false);
        acc[2] = __builtin_amdgcn_wmma_f32_16x16x32_bf16(false, a2.v, false, bm.v,
                                                         (short)0, acc[2], false, false);
        acc[3] = __builtin_amdgcn_wmma_f32_16x16x32_bf16(false, a3.v, false, bm.v,
                                                         (short)0, acc[3], false, false);
    }

    const float bv = bias[n0 + l15];
    if (mode == 2) {
        float* Y = (float*)Yraw;
#pragma unroll
        for (int sub = 0; sub < 4; ++sub)
#pragma unroll
            for (int r = 0; r < 8; ++r) {
                const int row = m0 + sub * 16 + r + (g << 3);
                Y[(size_t)row * N + n0 + l15] = acc[sub][r] + bv;
            }
    } else if (mode == 0) {
        unsigned short* Y = (unsigned short*)Yraw;
#pragma unroll
        for (int sub = 0; sub < 4; ++sub)
#pragma unroll
            for (int r = 0; r < 8; ++r) {
                const int row = m0 + sub * 16 + r + (g << 3);
                Y[(size_t)row * N + n0 + l15] = f2bfu(acc[sub][r] + bv);
            }
    } else {
        // transposed per-head store: Vt[((b*NHEAD + h)*HD + dh)*S + s]
        unsigned short* Y = (unsigned short*)Yraw;
        const int h  = n0 >> 6;
        const int dh = (n0 & 63) + l15;
#pragma unroll
        for (int sub = 0; sub < 4; ++sub)
#pragma unroll
            for (int r = 0; r < 8; ++r) {
                const int row = m0 + sub * 16 + r + (g << 3);   // = b*S + s
                const int bb = row >> 11;
                const int s  = row & (S_LEN - 1);
                Y[((size_t)(bb * NHEAD + h) * HD + dh) * S_LEN + s] =
                    f2bfu(acc[sub][r] + bv);
            }
    }
}

// ---------------------------------------------------------------------------
// Flash attention. One wave = 16 queries of one (b,h); 4 waves/block share the
// same (b,h) and key loop. K tiles (32x64 bf16) are DMA'd into LDS by the TDM
// (wave 0 issues, double-buffered, s_wait_tensorcnt 1 completes the current
// tile while the next streams). Scores: 4 WMMAs from LDS K; online softmax;
// P re-laid C->A through LDS; P@V: 4 WMMAs with V from transposed Vt staging.
// ---------------------------------------------------------------------------
__global__ __launch_bounds__(128)
void k_flash_attn(const unsigned short* __restrict__ Qb,
                  const unsigned short* __restrict__ Kb,
                  const unsigned short* __restrict__ Vt,
                  unsigned short* __restrict__ Ab) {
    __shared__ unsigned short lds_all[LDSU_K1 + 2048];   // P + 2x K staging

    const int lane = threadIdx.x & 31;
    const int wid  = threadIdx.x >> 5;
    const int g    = lane >> 4;
    const int l15  = lane & 15;
    const int h    = blockIdx.y;
    const int b    = blockIdx.z;
    const int q0   = blockIdx.x * 64 + wid * 16;

    // --- Q fragments (A layout), Dh=64 -> two K=32 chunks -------------------
    BF16x16 qa0, qa1;
    {
        const unsigned short* qrow =
            Qb + ((size_t)(b * S_LEN + q0 + l15)) * DM + (size_t)h * HD;
        qa0.q[0] = *(const u32x4*)(qrow + (g << 3));
        qa0.q[1] = *(const u32x4*)(qrow + (g << 3) + 16);
        qa1.q[0] = *(const u32x4*)(qrow + 32 + (g << 3));
        qa1.q[1] = *(const u32x4*)(qrow + 32 + (g << 3) + 16);
    }

    v8f o0 = {}, o1 = {}, o2 = {}, o3 = {};
    float mrow[8], lrow[8];
#pragma unroll
    for (int r = 0; r < 8; ++r) { mrow[r] = -1e30f; lrow[r] = 0.f; }

    unsigned short* pl = lds_all + LDSU_P + wid * 16 * LDS_P;
    const unsigned short* vhead = Vt + (size_t)(b * NHEAD + h) * HD * S_LEN;
    const unsigned short* kgbase = Kb + (size_t)(b * S_LEN) * DM + (size_t)h * HD;

    // --- TDM preamble: stream first K tile into buf0 ------------------------
    if (wid == 0) tdm_load_k_tile(kgbase, LDSB_K0);

    for (int it = 0; it < S_LEN / 32; ++it) {
        const int kb0 = it * 32;
        if (wid == 0) {
            // issue next tile (or dummy reload of tile 0 on the last trip)
            const int nkb = (kb0 + 32 < S_LEN) ? (kb0 + 32) : 0;
            tdm_load_k_tile(kgbase + (size_t)nkb * DM,
                            (it & 1) ? LDSB_K0 : LDSB_K1);
            __builtin_amdgcn_s_wait_tensorcnt(1);   // current tile complete
        }
        __syncthreads();                            // publish K tile to all waves

        const unsigned short* kst = lds_all + ((it & 1) ? LDSU_K1 : LDSU_K0);

        // ---- scores: S = Q @ K^T (two 16x16 key tiles) from LDS ------------
        v8f s0 = {}, s1 = {};
        {
            BF16x16 k00, k01, k10, k11;             // B layout: interleaved halves, as Q
            const unsigned short* kr0 = kst + l15 * 64;
            const unsigned short* kr1 = kst + (16 + l15) * 64;
            k00.q[0] = *(const u32x4*)(kr0 + (g << 3));
            k00.q[1] = *(const u32x4*)(kr0 + (g << 3) + 16);
            k01.q[0] = *(const u32x4*)(kr0 + 32 + (g << 3));
            k01.q[1] = *(const u32x4*)(kr0 + 32 + (g << 3) + 16);
            k10.q[0] = *(const u32x4*)(kr1 + (g << 3));
            k10.q[1] = *(const u32x4*)(kr1 + (g << 3) + 16);
            k11.q[0] = *(const u32x4*)(kr1 + 32 + (g << 3));
            k11.q[1] = *(const u32x4*)(kr1 + 32 + (g << 3) + 16);
            s0 = __builtin_amdgcn_wmma_f32_16x16x32_bf16(false, qa0.v, false, k00.v,
                                                         (short)0, s0, false, false);
            s0 = __builtin_amdgcn_wmma_f32_16x16x32_bf16(false, qa1.v, false, k01.v,
                                                         (short)0, s0, false, false);
            s1 = __builtin_amdgcn_wmma_f32_16x16x32_bf16(false, qa0.v, false, k10.v,
                                                         (short)0, s1, false, false);
            s1 = __builtin_amdgcn_wmma_f32_16x16x32_bf16(false, qa1.v, false, k11.v,
                                                         (short)0, s1, false, false);
        }

        // ---- online softmax (C layout: row = r + 8g, col = lane&15) --------
#pragma unroll
        for (int r = 0; r < 8; ++r) {
            float x0 = s0[r] * 0.125f;   // 1/sqrt(64)
            float x1 = s1[r] * 0.125f;
            float t = fmaxf(x0, x1);
            t = fmaxf(t, __shfl_xor(t, 1, 16));
            t = fmaxf(t, __shfl_xor(t, 2, 16));
            t = fmaxf(t, __shfl_xor(t, 4, 16));
            t = fmaxf(t, __shfl_xor(t, 8, 16));
            const float mn = fmaxf(mrow[r], t);
            const float alpha = __expf(mrow[r] - mn);
            mrow[r] = mn;
            const float p0 = __expf(x0 - mn);
            const float p1 = __expf(x1 - mn);
            float rs = p0 + p1;
            rs += __shfl_xor(rs, 1, 16);
            rs += __shfl_xor(rs, 2, 16);
            rs += __shfl_xor(rs, 4, 16);
            rs += __shfl_xor(rs, 8, 16);
            lrow[r] = lrow[r] * alpha + rs;
            o0[r] *= alpha; o1[r] *= alpha; o2[r] *= alpha; o3[r] *= alpha;

            const int row = r + (g << 3);
            pl[row * LDS_P + l15]      = f2bfu(p0);
            pl[row * LDS_P + 16 + l15] = f2bfu(p1);
        }
        asm volatile("s_wait_dscnt 0" ::: "memory");

        // ---- re-read P in A layout from LDS (2x ds_load_b128) --------------
        BF16x16 pa;
        pa.q[0] = *(const u32x4*)&pl[l15 * LDS_P + (g << 4)];
        pa.q[1] = *(const u32x4*)&pl[l15 * LDS_P + (g << 4) + 8];

        // ---- V fragments from transposed Vt: contiguous along keys ---------
        BF16x16 vf0, vf1, vf2, vf3;
        {
            const unsigned short* vr = vhead + (size_t)l15 * S_LEN + kb0 + (g << 4);
            const size_t fstride = (size_t)16 * S_LEN;
            vf0.q[0] = *(const u32x4*)(vr);
            vf0.q[1] = *(const u32x4*)(vr + 8);
            vf1.q[0] = *(const u32x4*)(vr + fstride);
            vf1.q[1] = *(const u32x4*)(vr + fstride + 8);
            vf2.q[0] = *(const u32x4*)(vr + 2 * fstride);
            vf2.q[1] = *(const u32x4*)(vr + 2 * fstride + 8);
            vf3.q[0] = *(const u32x4*)(vr + 3 * fstride);
            vf3.q[1] = *(const u32x4*)(vr + 3 * fstride + 8);
        }
        o0 = __builtin_amdgcn_wmma_f32_16x16x32_bf16(false, pa.v, false, vf0.v,
                                                     (short)0, o0, false, false);
        o1 = __builtin_amdgcn_wmma_f32_16x16x32_bf16(false, pa.v, false, vf1.v,
                                                     (short)0, o1, false, false);
        o2 = __builtin_amdgcn_wmma_f32_16x16x32_bf16(false, pa.v, false, vf2.v,
                                                     (short)0, o2, false, false);
        o3 = __builtin_amdgcn_wmma_f32_16x16x32_bf16(false, pa.v, false, vf3.v,
                                                     (short)0, o3, false, false);

        __syncthreads();   // all waves done with this K buffer before reuse
    }

    if (wid == 0) __builtin_amdgcn_s_wait_tensorcnt(0);   // drain dummy TDM

    // ---- normalize and store attention output (bf16, [B,S,D]) -------------
#pragma unroll
    for (int r = 0; r < 8; ++r) {
        const float inv = 1.0f / lrow[r];
        const int row = q0 + r + (g << 3);
        unsigned short* outp =
            Ab + ((size_t)(b * S_LEN + row)) * DM + (size_t)h * HD;
        outp[l15]      = f2bfu(o0[r] * inv);
        outp[16 + l15] = f2bfu(o1[r] * inv);
        outp[32 + l15] = f2bfu(o2[r] * inv);
        outp[48 + l15] = f2bfu(o3[r] * inv);
    }
}

// ---------------------------------------------------------------------------
extern "C" void kernel_launch(void* const* d_in, const int* in_sizes, int n_in,
                              void* d_out, int out_size, void* d_ws, size_t ws_size,
                              hipStream_t stream) {
    (void)in_sizes; (void)n_in; (void)out_size; (void)ws_size;

    const float* q  = (const float*)d_in[0];
    const float* k  = (const float*)d_in[1];
    const float* v  = (const float*)d_in[2];
    const float* Wq = (const float*)d_in[3];
    const float* bq = (const float*)d_in[4];
    const float* Wk = (const float*)d_in[5];
    const float* bk = (const float*)d_in[6];
    const float* Wv = (const float*)d_in[7];
    const float* bv = (const float*)d_in[8];
    const float* Wo = (const float*)d_in[9];
    const float* bo = (const float*)d_in[10];
    float* out = (float*)d_out;

    const int M = BATCH * S_LEN;                  // 8192
    const size_t MN = (size_t)M * DM;             // 8.39M elems
    const size_t WN = (size_t)DM * DM;            // 1.05M elems

    unsigned short* p   = (unsigned short*)d_ws;  // ~126 MB staging (L2-resident)
    unsigned short* qf  = p;                      // bf16 inputs
    unsigned short* kf  = p + MN;
    unsigned short* vf  = p + 2 * MN;
    unsigned short* Wqb = p + 3 * MN;             // bf16 weights
    unsigned short* Wkb = Wqb + WN;
    unsigned short* Wvb = Wqb + 2 * WN;
    unsigned short* Wob = Wqb + 3 * WN;
    unsigned short* Qb  = Wqb + 4 * WN;           // projected Q,K
    unsigned short* Kb  = Qb + MN;
    unsigned short* Vt  = Qb + 2 * MN;            // V transposed per head
    unsigned short* Ab  = Qb + 3 * MN;            // attention output

    // --- stage everything as bf16 ------------------------------------------
    const int gin = (int)(MN / 8 / 256);          // 4096 blocks
    const int gw  = (int)(WN / 8 / 256);          // 512 blocks
    k_cvt_bf16<<<gin, 256, 0, stream>>>(q,  qf,  (int)MN);
    k_cvt_bf16<<<gin, 256, 0, stream>>>(k,  kf,  (int)MN);
    k_cvt_bf16<<<gin, 256, 0, stream>>>(v,  vf,  (int)MN);
    k_cvt_bf16<<<gw,  256, 0, stream>>>(Wq, Wqb, (int)WN);
    k_cvt_bf16<<<gw,  256, 0, stream>>>(Wk, Wkb, (int)WN);
    k_cvt_bf16<<<gw,  256, 0, stream>>>(Wv, Wvb, (int)WN);
    k_cvt_bf16<<<gw,  256, 0, stream>>>(Wo, Wob, (int)WN);

    // --- projections (64x16 tile per wave, 8 waves per block) --------------
    const int tiles = (M / 64) * (DM / 16);       // 8192
    dim3 gproj(tiles / 8);                        // 1024 blocks
    k_gemm_bf16<<<gproj, 256, 0, stream>>>(qf, Wqb, bq, Qb, M, DM, DM, 0);
    k_gemm_bf16<<<gproj, 256, 0, stream>>>(kf, Wkb, bk, Kb, M, DM, DM, 0);
    k_gemm_bf16<<<gproj, 256, 0, stream>>>(vf, Wvb, bv, Vt, M, DM, DM, 1);

    // --- attention (TDM-fed K tiles) ----------------------------------------
    dim3 gattn(S_LEN / 64, NHEAD, BATCH);         // 16 queries/wave, 4 waves/blk
    k_flash_attn<<<gattn, 128, 0, stream>>>(Qb, Kb, Vt, Ab);

    // --- output projection (f32 out) ----------------------------------------
    k_gemm_bf16<<<gproj, 256, 0, stream>>>(Ab, Wob, bo, out, M, DM, DM, 2);
}
